// GatP3_17437567221935
// MI455X (gfx1250) — compile-verified, matrix operand order checked by reading, not run on hardware
//
#include <hip/hip_runtime.h>
#include <hip/hip_bf16.h>
#include <cmath>

#define NN 100000
#define NE 1600000

typedef __attribute__((ext_vector_type(16))) __bf16 v16bf;
typedef __attribute__((ext_vector_type(8)))  float  v8f;

// ---------------- helpers ----------------

__device__ __forceinline__ float atomic_max_f32(float* addr, float val) {
    if (val >= 0.0f) {
        return __int_as_float(atomicMax((int*)addr, __float_as_int(val)));
    } else {
        return __uint_as_float(atomicMin((unsigned int*)addr, __float_as_uint(val)));
    }
}

__global__ void fill_f32(float* __restrict__ p, float v, int n) {
    int i = blockIdx.x * blockDim.x + threadIdx.x;
    if (i < n) p[i] = v;
}

__global__ void f32_to_bf16(const float* __restrict__ in, __bf16* __restrict__ out, int n) {
    int i = blockIdx.x * blockDim.x + threadIdx.x;
    if (i < n) out[i] = (__bf16)in[i];
}

// W: [C, O] row-major -> Wt: [O, C] bf16 (so B fragments load contiguous K)
__global__ void transpose_w_bf16(const float* __restrict__ W, __bf16* __restrict__ Wt,
                                 int C, int O) {
    int i = blockIdx.x * blockDim.x + threadIdx.x;
    if (i < C * O) {
        int c = i / O, o = i - c * O;
        Wt[o * C + c] = (__bf16)W[i];
    }
}

// ---------------- WMMA GEMM: C[M,N] = A[M,K] * Bt[N,K]^T ----------------
// M % 32 == 0, N % 64 == 0, K compile-time (% 32 == 0).
// One wave -> 32x64 output tile. Per k-step: load a0,a1,b0..b3 (12 x b128)
// up front, then 8 WMMAs, so the scheduler can clause loads and overlap with
// the previous step's matrix ops.
template<int K>
__global__ void __launch_bounds__(256)
gemm_bf16_wmma(const __bf16* __restrict__ A, const __bf16* __restrict__ Bt,
               float* __restrict__ C, int M, int N) {
    const int lane = threadIdx.x & 31;
    const int wid  = (blockIdx.x * blockDim.x + threadIdx.x) >> 5;
    const int ngroups = N >> 6;
    const int total   = (M >> 5) * ngroups;
    if (wid >= total) return;               // uniform per wave; EXEC full inside

    const int mtile = wid / ngroups;
    const int ng    = wid - mtile * ngroups;
    const int row0  = mtile << 5;
    const int col0  = ng << 6;
    const int half  = lane >> 4;            // 0: K 0..15, 1: K 16..31
    const int r     = lane & 15;

    const __bf16* aRow0 = A + (size_t)(row0 + r) * K + half * 16;
    const __bf16* aRow1 = aRow0 + (size_t)16 * K;
    const __bf16* bRow  = Bt + (size_t)(col0 + r) * K + half * 16;

    v8f acc[2][4] = {};
#pragma unroll
    for (int k0 = 0; k0 < K; k0 += 32) {
        v16bf a0 = *(const v16bf*)(aRow0 + k0);
        v16bf a1 = *(const v16bf*)(aRow1 + k0);
        v16bf b0 = *(const v16bf*)(bRow + (size_t)0  * K + k0);
        v16bf b1 = *(const v16bf*)(bRow + (size_t)16 * K + k0);
        v16bf b2 = *(const v16bf*)(bRow + (size_t)32 * K + k0);
        v16bf b3 = *(const v16bf*)(bRow + (size_t)48 * K + k0);
        acc[0][0] = __builtin_amdgcn_wmma_f32_16x16x32_bf16(false, a0, false, b0, (short)0, acc[0][0], false, false);
        acc[1][0] = __builtin_amdgcn_wmma_f32_16x16x32_bf16(false, a1, false, b0, (short)0, acc[1][0], false, false);
        acc[0][1] = __builtin_amdgcn_wmma_f32_16x16x32_bf16(false, a0, false, b1, (short)0, acc[0][1], false, false);
        acc[1][1] = __builtin_amdgcn_wmma_f32_16x16x32_bf16(false, a1, false, b1, (short)0, acc[1][1], false, false);
        acc[0][2] = __builtin_amdgcn_wmma_f32_16x16x32_bf16(false, a0, false, b2, (short)0, acc[0][2], false, false);
        acc[1][2] = __builtin_amdgcn_wmma_f32_16x16x32_bf16(false, a1, false, b2, (short)0, acc[1][2], false, false);
        acc[0][3] = __builtin_amdgcn_wmma_f32_16x16x32_bf16(false, a0, false, b3, (short)0, acc[0][3], false, false);
        acc[1][3] = __builtin_amdgcn_wmma_f32_16x16x32_bf16(false, a1, false, b3, (short)0, acc[1][3], false, false);
    }

    // C/D layout: VGPR rr, lanes 0-15 -> row rr, lanes 16-31 -> row rr+8; col = lane&15
    const int col = lane & 15;
#pragma unroll
    for (int mh = 0; mh < 2; ++mh) {
        float* crow = C + (size_t)(row0 + mh * 16 + half * 8) * N + col0 + col;
#pragma unroll
        for (int j = 0; j < 4; ++j) {
#pragma unroll
            for (int rr = 0; rr < 8; ++rr) {
                crow[(size_t)rr * N + j * 16] = acc[mh][j][rr];
            }
        }
    }
}

// ---------------- attention logits per node ----------------
// h: [N, H*D]; al/ar: [H, D]; el/er: [N, H]
__global__ void el_er_kernel(const float* __restrict__ h, const float* __restrict__ al,
                             const float* __restrict__ ar, float* __restrict__ el,
                             float* __restrict__ er, int N, int H, int D) {
    int i = blockIdx.x * blockDim.x + threadIdx.x;
    if (i >= N * H) return;
    int n = i / H, hh = i - n * H;
    const float* hv  = h  + (size_t)n * H * D + hh * D;
    const float* alv = al + hh * D;
    const float* arv = ar + hh * D;
    float sl = 0.f, sr = 0.f;
    for (int d = 0; d < D; ++d) { float v = hv[d]; sl += v * alv[d]; sr += v * arv[d]; }
    el[i] = sl; er[i] = sr;
}

__device__ __forceinline__ float leaky02(float v) { return v > 0.f ? v : 0.2f * v; }

// pass 1: segment max over dst
__global__ void edge_max(const long long* __restrict__ src, const long long* __restrict__ dst,
                         const float* __restrict__ el, const float* __restrict__ er,
                         float* __restrict__ m, int E, int H) {
    int i = blockIdx.x * blockDim.x + threadIdx.x;
    if (i >= E * H) return;
    int e = i / H, hh = i - e * H;
    int s = (int)src[e], t = (int)dst[e];
    float v = leaky02(el[s * H + hh] + er[t * H + hh]);
    atomic_max_f32(&m[t * H + hh], v);
}

// pass 2: exp numerators + segment sum
__global__ void edge_exp(const long long* __restrict__ src, const long long* __restrict__ dst,
                         const float* __restrict__ el, const float* __restrict__ er,
                         const float* __restrict__ m, float* __restrict__ ex,
                         float* __restrict__ den, int E, int H) {
    int i = blockIdx.x * blockDim.x + threadIdx.x;
    if (i >= E * H) return;
    int e = i / H, hh = i - e * H;
    int s = (int)src[e], t = (int)dst[e];
    float v  = leaky02(el[s * H + hh] + er[t * H + hh]);
    float xv = __expf(v - m[t * H + hh]);
    ex[i] = xv;
    atomicAdd(&den[t * H + hh], xv);
}

// pass 3: acc[dst] += alpha * h[src]; one wave per edge, contiguous F/32 floats per lane
template<int H, int D>
__global__ void edge_agg(const long long* __restrict__ src, const long long* __restrict__ dst,
                         const float* __restrict__ h, const float* __restrict__ ex,
                         const float* __restrict__ den, float* __restrict__ acc, int E) {
    const int lane = threadIdx.x & 31;
    int e = (blockIdx.x * blockDim.x + threadIdx.x) >> 5;
    if (e >= E) return;
    int s = (int)src[e], t = (int)dst[e];
    constexpr int F   = H * D;
    constexpr int PER = F / 32;
    const float* hs = h + (size_t)s * F;
    float* at = acc + (size_t)t * F;
    __builtin_prefetch(hs + lane * PER, 0, 0);
#pragma unroll
    for (int j = 0; j < PER; ++j) {
        int f  = lane * PER + j;
        int hh = f / D;
        float alpha = ex[(size_t)e * H + hh] / den[t * H + hh];
        atomicAdd(&at[f], alpha * hs[f]);
    }
}

// out_bf16 = bf16(relu(acc + b))  (feeds layer-2 GEMM)
__global__ void relu_bias_bf16(const float* __restrict__ acc, const float* __restrict__ b,
                               __bf16* __restrict__ out, int N, int F) {
    int i = blockIdx.x * blockDim.x + threadIdx.x;
    if (i >= N * F) return;
    int f = i % F;
    float v = acc[i] + b[f];
    out[i] = (__bf16)(v > 0.f ? v : 0.f);
}

__global__ void add_bias(float* __restrict__ out, const float* __restrict__ b, int N, int F) {
    int i = blockIdx.x * blockDim.x + threadIdx.x;
    if (i >= N * F) return;
    out[i] += b[i % F];
}

// ---------------- launch ----------------

static inline int gs(long long n, int b) { return (int)((n + b - 1) / b); }

extern "C" void kernel_launch(void* const* d_in, const int* in_sizes, int n_in,
                              void* d_out, int out_size, void* d_ws, size_t ws_size,
                              hipStream_t stream) {
    const float*     feat = (const float*)d_in[0];
    const long long* src  = (const long long*)d_in[1];
    const long long* dst  = (const long long*)d_in[2];
    const float*     W1   = (const float*)d_in[3];   // [256,4,64]
    const float*     al1  = (const float*)d_in[4];   // [4,64]
    const float*     ar1  = (const float*)d_in[5];
    const float*     b1   = (const float*)d_in[6];   // [256]
    const float*     W2   = (const float*)d_in[7];   // [256,1,64]
    const float*     al2  = (const float*)d_in[8];   // [1,64]
    const float*     ar2  = (const float*)d_in[9];
    const float*     b2   = (const float*)d_in[10];  // [64]
    float* out = (float*)d_out;                      // [N,64]

    const int N = NN, E = NE, C = 256, H1 = 4, D = 64, F1 = 256, F2 = 64;
    const int B = 256;

    // workspace carve-out (256B aligned)
    char* ws = (char*)d_ws;
    size_t off = 0;
    auto carve = [&](size_t bytes) -> void* {
        void* p = ws + off;
        off = (off + bytes + 255) & ~((size_t)255);
        return p;
    };
    __bf16* xbf  = (__bf16*)carve((size_t)N * C * 2);   // feat bf16, reused for layer-2 input
    __bf16* w1t  = (__bf16*)carve((size_t)F1 * C * 2);  // [256 out][256 in]
    __bf16* w2t  = (__bf16*)carve((size_t)F2 * C * 2);  // [64 out][256 in]
    float*  h1   = (float*)carve((size_t)N * F1 * 4);
    float*  acc1 = (float*)carve((size_t)N * F1 * 4);   // reused as h2 in layer 2
    float*  el   = (float*)carve((size_t)N * H1 * 4);   // reused for layer 2 (H=1)
    float*  er   = (float*)carve((size_t)N * H1 * 4);
    float*  m    = (float*)carve((size_t)N * H1 * 4);
    float*  den  = (float*)carve((size_t)N * H1 * 4);
    float*  ex   = (float*)carve((size_t)E * H1 * 4);   // reused for layer 2
    float*  h2   = acc1;

    const float NEG_INF = -INFINITY;

    // ---- prep: bf16 conversions ----
    transpose_w_bf16<<<gs((long long)C * F1, B), B, 0, stream>>>(W1, w1t, C, F1);
    transpose_w_bf16<<<gs((long long)C * F2, B), B, 0, stream>>>(W2, w2t, C, F2);
    f32_to_bf16<<<gs((long long)N * C, B), B, 0, stream>>>(feat, xbf, N * C);

    // ---- layer 1 ----
    {
        int waves = (N / 32) * (F1 / 64);
        gemm_bf16_wmma<256><<<gs(waves, B / 32), B, 0, stream>>>(xbf, w1t, h1, N, F1);
    }
    el_er_kernel<<<gs((long long)N * H1, B), B, 0, stream>>>(h1, al1, ar1, el, er, N, H1, D);
    fill_f32<<<gs((long long)N * H1, B), B, 0, stream>>>(m, NEG_INF, N * H1);
    fill_f32<<<gs((long long)N * H1, B), B, 0, stream>>>(den, 0.0f, N * H1);
    fill_f32<<<gs((long long)N * F1, B), B, 0, stream>>>(acc1, 0.0f, N * F1);
    edge_max<<<gs((long long)E * H1, B), B, 0, stream>>>(src, dst, el, er, m, E, H1);
    edge_exp<<<gs((long long)E * H1, B), B, 0, stream>>>(src, dst, el, er, m, ex, den, E, H1);
    edge_agg<4, 64><<<gs((long long)E * 32, B), B, 0, stream>>>(src, dst, h1, ex, den, acc1, E);
    relu_bias_bf16<<<gs((long long)N * F1, B), B, 0, stream>>>(acc1, b1, xbf, N, F1);

    // ---- layer 2 ----
    {
        int waves = (N / 32) * (F2 / 64);
        gemm_bf16_wmma<256><<<gs(waves, B / 32), B, 0, stream>>>(xbf, w2t, h2, N, F2);
    }
    el_er_kernel<<<gs((long long)N, B), B, 0, stream>>>(h2, al2, ar2, el, er, N, 1, D);
    fill_f32<<<gs((long long)N, B), B, 0, stream>>>(m, NEG_INF, N);
    fill_f32<<<gs((long long)N, B), B, 0, stream>>>(den, 0.0f, N);
    fill_f32<<<gs((long long)N * F2, B), B, 0, stream>>>(out, 0.0f, N * F2);
    edge_max<<<gs((long long)E, B), B, 0, stream>>>(src, dst, el, er, m, E, 1);
    edge_exp<<<gs((long long)E, B), B, 0, stream>>>(src, dst, el, er, m, ex, den, E, 1);
    edge_agg<1, 64><<<gs((long long)E * 32, B), B, 0, stream>>>(src, dst, h2, ex, den, out, E);
    add_bias<<<gs((long long)N * F2, B), B, 0, stream>>>(out, b2, N, F2);
}
